// Attention_85890755986067
// MI455X (gfx1250) — compile-verified, matrix operand order checked by reading, not run on hardware
//
#include <hip/hip_runtime.h>
#include <hip/hip_bf16.h>
#include <cstdint>

// ---------------------------------------------------------------------------
// Types for CDNA5 WMMA (gfx1250, wave32)
// ---------------------------------------------------------------------------
typedef __attribute__((ext_vector_type(16))) __bf16 bf16x16;  // A/B fragment (8 VGPRs)
typedef __attribute__((ext_vector_type(8)))  __bf16 bf16x8;   // 16-byte chunk
typedef __attribute__((ext_vector_type(8)))  float  f32x8;    // C/D fragment (8 VGPRs)

#define HEAD_DIM   128
#define NUM_HEADS  16
#define SEQ        4096
#define HID        2048
#define QKV_OUT    ((NUM_HEADS * 3) * HEAD_DIM)   // 6144

// ---------------------------------------------------------------------------
// WMMA helper: D = A(16x32 bf16) * B(32x16 bf16) + C(16x16 f32)
// ---------------------------------------------------------------------------
__device__ __forceinline__ f32x8 wmma_bf16(bf16x16 a, bf16x16 b, f32x8 c) {
    return __builtin_amdgcn_wmma_f32_16x16x32_bf16(
        /*neg_a=*/false, a, /*neg_b=*/false, b,
        /*c_mod=*/(short)0, c, /*reuse_a=*/false, /*reuse_b=*/false);
}

// ---------------------------------------------------------------------------
// Load a 16x32 bf16 A-fragment (or mirrored 32x16 B-fragment) from a
// K-contiguous row-major matrix.  ISA layout (16-bit A, 16x32):
//   lane L:  row = L&15 ; hi = L>>4
//   elems 0..7  hold K = hi*8 + 0..7 ; elems 8..15 hold K = 16 + hi*8 + 0..7
// ---------------------------------------------------------------------------
__device__ __forceinline__ bf16x16 load_frag(const __bf16* p, int ld, int lane) {
    const int r  = lane & 15;
    const int hi = (lane >> 4) & 1;
    const __bf16* row = p + (size_t)r * ld + hi * 8;
    bf16x8 lo = *(const bf16x8*)(row);        // K = hi*8 + 0..7
    bf16x8 hc = *(const bf16x8*)(row + 16);   // K = 16 + hi*8 + 0..7
    bf16x16 out;
#pragma unroll
    for (int i = 0; i < 8; ++i) { out[i] = lo[i]; out[i + 8] = hc[i]; }
    return out;
}

// ---------------------------------------------------------------------------
// CDNA5 async global->LDS copy (ASYNCcnt path).  LDS byte offset is the low
// 32 bits of the generic shared-pointer (ISA 10.2 LDS aperture: addr[31:0]).
// ---------------------------------------------------------------------------
__device__ __forceinline__ void async_copy_b128(const __bf16* gsrc, __bf16* ldst) {
    unsigned loff = (unsigned)(uintptr_t)ldst;
    unsigned long long gaddr = (unsigned long long)(uintptr_t)gsrc;
    asm volatile("global_load_async_to_lds_b128 %0, %1, off"
                 :: "v"(loff), "v"(gaddr) : "memory");
}
__device__ __forceinline__ void wait_async_zero() {
    asm volatile("s_wait_asynccnt 0x0" ::: "memory");
}

// ---------------------------------------------------------------------------
// Kernel 0: fp32 -> bf16 conversion
// ---------------------------------------------------------------------------
__global__ void cvt_bf16_kernel(const float* __restrict__ in,
                                __bf16* __restrict__ out, int n) {
    int i = blockIdx.x * 256 + threadIdx.x;
    if (i < n) out[i] = (__bf16)in[i];
}

// ---------------------------------------------------------------------------
// Kernel 1: QKV GEMM.  C[m,n] = sum_k hidden[m,k] * w_qkv[n,k]
//   M=4096, N=6144, K=2048.  WG = 256 thr = 8 waves, WG tile 64x128,
//   wave tile 32x32 (2x2 WMMA).  Epilogue scatters into Q[h][s][d],
//   K[h][s][d] and V^T[h][d][s] (bf16).
// ---------------------------------------------------------------------------
__global__ __launch_bounds__(256)
void qkv_gemm_kernel(const __bf16* __restrict__ A, const __bf16* __restrict__ B,
                     __bf16* __restrict__ qbuf, __bf16* __restrict__ kbuf,
                     __bf16* __restrict__ vT) {
    const int lane = threadIdx.x & 31;
    const int wave = threadIdx.x >> 5;
    const int wm = wave & 1;          // 2 row-tiles of 32
    const int wn = wave >> 1;         // 4 col-tiles of 32
    const int m0 = blockIdx.x * 64 + wm * 32;
    const int n0 = blockIdx.y * 128 + wn * 32;

    f32x8 acc00 = {}, acc01 = {}, acc10 = {}, acc11 = {};

    for (int k0 = 0; k0 < HID; k0 += 32) {
        bf16x16 a0 = load_frag(A + (size_t)m0 * HID + k0, HID, lane);
        bf16x16 a1 = load_frag(A + (size_t)(m0 + 16) * HID + k0, HID, lane);
        bf16x16 b0 = load_frag(B + (size_t)n0 * HID + k0, HID, lane);
        bf16x16 b1 = load_frag(B + (size_t)(n0 + 16) * HID + k0, HID, lane);
        acc00 = wmma_bf16(a0, b0, acc00);
        acc01 = wmma_bf16(a0, b1, acc01);
        acc10 = wmma_bf16(a1, b0, acc10);
        acc11 = wmma_bf16(a1, b1, acc11);
    }

    const int hi = lane >> 4;
    const int nl = lane & 15;
#pragma unroll
    for (int tm = 0; tm < 2; ++tm) {
#pragma unroll
        for (int tn = 0; tn < 2; ++tn) {
            const f32x8 acc = (tm == 0) ? (tn == 0 ? acc00 : acc01)
                                        : (tn == 0 ? acc10 : acc11);
            const int mbase = m0 + tm * 16;
            const int nbase = n0 + tn * 16;
#pragma unroll
            for (int i = 0; i < 8; ++i) {
                const int m = mbase + i + hi * 8;       // sequence position
                const int n = nbase + nl;               // qkv output channel
                const __bf16 v = (__bf16)acc[i];
                if (n < NUM_HEADS * HEAD_DIM) {
                    const int h = n >> 7, d = n & 127;
                    qbuf[((size_t)h * SEQ + m) * HEAD_DIM + d] = v;
                } else if (n < 2 * NUM_HEADS * HEAD_DIM) {
                    const int h = (n - NUM_HEADS * HEAD_DIM) >> 7, d = n & 127;
                    kbuf[((size_t)h * SEQ + m) * HEAD_DIM + d] = v;
                } else {
                    const int h = (n - 2 * NUM_HEADS * HEAD_DIM) >> 7, d = n & 127;
                    vT[((size_t)h * HEAD_DIM + d) * SEQ + m] = v;   // transposed
                }
            }
        }
    }
}

// ---------------------------------------------------------------------------
// Kernel 2: RoPE applied in place to Q and K (bf16, fp32 math).
// ---------------------------------------------------------------------------
__global__ void rope_kernel(__bf16* __restrict__ qbuf, __bf16* __restrict__ kbuf) {
    const int idx = blockIdx.x * 256 + threadIdx.x;
    if (idx >= SEQ * NUM_HEADS * (HEAD_DIM / 2)) return;
    const int i = idx & 63;
    const int t = idx >> 6;
    const int h = t & (NUM_HEADS - 1);
    const int s = t >> 4;
    const float inv = __expf(-(float)i * 0.14391156831213f);  // 10000^{-i/64}
    const float ang = (float)s * inv;
    const float c = __cosf(ang), sn = __sinf(ang);
    const size_t base = ((size_t)h * SEQ + s) * HEAD_DIM;
    {
        float xr = (float)qbuf[base + i], xi = (float)qbuf[base + 64 + i];
        qbuf[base + i]      = (__bf16)(xr * c - xi * sn);
        qbuf[base + 64 + i] = (__bf16)(xi * c + xr * sn);
    }
    {
        float xr = (float)kbuf[base + i], xi = (float)kbuf[base + 64 + i];
        kbuf[base + i]      = (__bf16)(xr * c - xi * sn);
        kbuf[base + 64 + i] = (__bf16)(xi * c + xr * sn);
    }
}

// ---------------------------------------------------------------------------
// Kernel 3: causal flash attention, WG-cooperative LDS staging.
//   One WG = (head, 128-row q block); 8 waves, each owns a 16-row q tile.
//   Per key iteration (32 keys):
//     * all 256 threads async-copy K tile (32x128) and V^T tile (128x32)
//       into LDS with global_load_async_to_lds_b128 (ASYNCcnt),
//       s_wait_asynccnt + barrier, then
//     * each wave: 8 QK^T WMMAs from LDS, fp32 online softmax,
//       P staged through per-wave LDS (s_wait_dscnt), 8 PV WMMAs.
//   All waves run the block-uniform causal trip count; waves fully in the
//   future of a key tile skip the compute body (barriers stay uniform).
// ---------------------------------------------------------------------------
__global__ __launch_bounds__(256)
void attn_kernel(const __bf16* __restrict__ qbuf, const __bf16* __restrict__ kbuf,
                 const __bf16* __restrict__ vT, __bf16* __restrict__ attn) {
    __shared__ __align__(16) __bf16 kTile[32 * HEAD_DIM];    // 32 keys x 128 d
    __shared__ __align__(16) __bf16 vTile[HEAD_DIM * 32];    // 128 d x 32 keys
    __shared__ __align__(16) __bf16 lds_p[8][16 * 32];       // per-wave P tile

    const int lane = threadIdx.x & 31;
    const int wave = threadIdx.x >> 5;
    const int h    = blockIdx.x >> 5;          // 16 heads
    const int qblk = blockIdx.x & 31;          // 32 blocks of 128 q rows
    const int qb0  = qblk * 128;
    const int q0   = qb0 + wave * 16;          // this wave's 16 q rows
    const int hi = lane >> 4;
    const int nl = lane & 15;
    const float scale = 0.08838834764831845f;  // 1/sqrt(128)
    const float NEGINF = -1e30f;

    // Q A-fragments: 16 rows x 128 dims = 4 chunks of K=32
    bf16x16 qf[4];
#pragma unroll
    for (int c = 0; c < 4; ++c)
        qf[c] = load_frag(qbuf + ((size_t)h * SEQ + q0) * HEAD_DIM + c * 32,
                          HEAD_DIM, lane);

    f32x8 o[8];
#pragma unroll
    for (int dt = 0; dt < 8; ++dt) o[dt] = (f32x8){};
    float mrow[8], lrow[8];
#pragma unroll
    for (int i = 0; i < 8; ++i) { mrow[i] = NEGINF; lrow[i] = 0.f; }

    __bf16* myp = &lds_p[wave][0];
    const int niter = (qb0 + 128) / 32;        // uniform across the WG

    const __bf16* kbase = kbuf + (size_t)h * SEQ * HEAD_DIM;
    const __bf16* vbase = vT + (size_t)h * HEAD_DIM * SEQ;
    const int tid = threadIdx.x;

    for (int it = 0; it < niter; ++it) {
        const int j0 = it * 32;

        // protect tiles still being read by other waves from the last iter
        __syncthreads();

        // ---- cooperative async staging: K tile (512 x b128 chunks) ----
        {
            // chunks c = tid*2, tid*2+1 ; row = c>>4 ; col-chunk = c&15
            const int c0 = tid * 2;
#pragma unroll
            for (int u = 0; u < 2; ++u) {
                const int c = c0 + u;
                const int row = c >> 4;
                const int cc  = c & 15;
                async_copy_b128(kbase + (size_t)(j0 + row) * HEAD_DIM + cc * 8,
                                &kTile[row * HEAD_DIM + cc * 8]);
            }
            // V^T tile: 512 chunks ; d = c>>2 ; part = c&3
#pragma unroll
            for (int u = 0; u < 2; ++u) {
                const int c = c0 + u;
                const int d = c >> 2;
                const int part = c & 3;
                async_copy_b128(vbase + (size_t)d * SEQ + j0 + part * 8,
                                &vTile[d * 32 + part * 8]);
            }
        }
        wait_async_zero();        // our own ASYNCcnt -> 0
        __syncthreads();          // whole tile visible to all waves

        // waves entirely in the future of this key tile: nothing to add
        if (j0 <= q0 + 15) {
            // ---- S = Q K^T from LDS (two 16x16 score tiles) ----
            f32x8 s0 = {}, s1 = {};
#pragma unroll
            for (int c = 0; c < 4; ++c) {
                bf16x16 kf0 = load_frag(&kTile[0 * HEAD_DIM + c * 32], HEAD_DIM, lane);
                bf16x16 kf1 = load_frag(&kTile[16 * HEAD_DIM + c * 32], HEAD_DIM, lane);
                s0 = wmma_bf16(qf[c], kf0, s0);
                s1 = wmma_bf16(qf[c], kf1, s1);
            }

            // ---- online softmax over the 32 new columns ----
            float alpha[8];
#pragma unroll
            for (int i = 0; i < 8; ++i) {
                const int qrow = q0 + i + hi * 8;
                float x0 = (j0 + nl      <= qrow) ? s0[i] * scale : NEGINF;
                float x1 = (j0 + 16 + nl <= qrow) ? s1[i] * scale : NEGINF;
                float mx = fmaxf(x0, x1);
#pragma unroll
                for (int off = 1; off < 16; off <<= 1)
                    mx = fmaxf(mx, __shfl_xor(mx, off, 32));
                const float mnew = fmaxf(mrow[i], mx);
                alpha[i] = __expf(mrow[i] - mnew);
                const float p0 = __expf(x0 - mnew);
                const float p1 = __expf(x1 - mnew);
                float rs = p0 + p1;
#pragma unroll
                for (int off = 1; off < 16; off <<= 1)
                    rs += __shfl_xor(rs, off, 32);
                lrow[i] = lrow[i] * alpha[i] + rs;
                mrow[i] = mnew;
                // stage P (Cfrag -> Afrag transpose) in this wave's LDS tile
                myp[(i + hi * 8) * 32 + nl]      = (__bf16)p0;
                myp[(i + hi * 8) * 32 + 16 + nl] = (__bf16)p1;
            }

#pragma unroll
            for (int dt = 0; dt < 8; ++dt)
#pragma unroll
                for (int i = 0; i < 8; ++i) o[dt][i] *= alpha[i];

            // per-wave DS ordering: wait for our own P stores only
            asm volatile("s_wait_dscnt 0x0" ::: "memory");

            // ---- O += P V from LDS ----
            bf16x16 pa = load_frag(myp, 32, lane);
#pragma unroll
            for (int dt = 0; dt < 8; ++dt) {
                bf16x16 vb = load_frag(&vTile[dt * 16 * 32], 32, lane);
                o[dt] = wmma_bf16(pa, vb, o[dt]);
            }
        }
    }

    // ---- normalize and store attn[s][h*128+d] (bf16) ----
#pragma unroll
    for (int dt = 0; dt < 8; ++dt) {
#pragma unroll
        for (int i = 0; i < 8; ++i) {
            const int m = q0 + i + hi * 8;
            const float val = o[dt][i] / lrow[i];
            attn[(size_t)m * (NUM_HEADS * HEAD_DIM) + h * HEAD_DIM + dt * 16 + nl]
                = (__bf16)val;
        }
    }
}

// ---------------------------------------------------------------------------
// Kernel 4: output projection.  out[m,n] = sum_o attn[m,o] * w_o[n,o]
// ---------------------------------------------------------------------------
__global__ __launch_bounds__(256)
void out_gemm_kernel(const __bf16* __restrict__ A, const __bf16* __restrict__ B,
                     float* __restrict__ out) {
    const int lane = threadIdx.x & 31;
    const int wave = threadIdx.x >> 5;
    const int wm = wave & 1;
    const int wn = wave >> 1;
    const int m0 = blockIdx.x * 64 + wm * 32;
    const int n0 = blockIdx.y * 128 + wn * 32;

    f32x8 acc00 = {}, acc01 = {}, acc10 = {}, acc11 = {};

    for (int k0 = 0; k0 < HID; k0 += 32) {
        bf16x16 a0 = load_frag(A + (size_t)m0 * HID + k0, HID, lane);
        bf16x16 a1 = load_frag(A + (size_t)(m0 + 16) * HID + k0, HID, lane);
        bf16x16 b0 = load_frag(B + (size_t)n0 * HID + k0, HID, lane);
        bf16x16 b1 = load_frag(B + (size_t)(n0 + 16) * HID + k0, HID, lane);
        acc00 = wmma_bf16(a0, b0, acc00);
        acc01 = wmma_bf16(a0, b1, acc01);
        acc10 = wmma_bf16(a1, b0, acc10);
        acc11 = wmma_bf16(a1, b1, acc11);
    }

    const int hi = lane >> 4;
    const int nl = lane & 15;
#pragma unroll
    for (int tm = 0; tm < 2; ++tm) {
#pragma unroll
        for (int tn = 0; tn < 2; ++tn) {
            const f32x8 acc = (tm == 0) ? (tn == 0 ? acc00 : acc01)
                                        : (tn == 0 ? acc10 : acc11);
#pragma unroll
            for (int i = 0; i < 8; ++i) {
                const int m = m0 + tm * 16 + i + hi * 8;
                const int n = n0 + tn * 16 + nl;
                out[(size_t)m * HID + n] = acc[i];
            }
        }
    }
}

// ---------------------------------------------------------------------------
// Host-side launcher
// ---------------------------------------------------------------------------
extern "C" void kernel_launch(void* const* d_in, const int* in_sizes, int n_in,
                              void* d_out, int out_size, void* d_ws, size_t ws_size,
                              hipStream_t stream) {
    const float* hidden = (const float*)d_in[0];   // [1,4096,2048]
    const float* w_qkv  = (const float*)d_in[1];   // [6144,2048]
    const float* w_o    = (const float*)d_in[2];   // [2048,2048]
    float* out = (float*)d_out;                    // [1,4096,2048] fp32

    __bf16* ws  = (__bf16*)d_ws;
    __bf16* Abf = ws;                                      // 4096*2048
    __bf16* Wq  = Abf + (size_t)SEQ * HID;                 // 6144*2048
    __bf16* Wo  = Wq + (size_t)QKV_OUT * HID;              // 2048*2048
    __bf16* qb  = Wo + (size_t)HID * HID;                  // 16*4096*128
    __bf16* kb  = qb + (size_t)NUM_HEADS * SEQ * HEAD_DIM;
    __bf16* vT  = kb + (size_t)NUM_HEADS * SEQ * HEAD_DIM;
    __bf16* atn = vT + (size_t)NUM_HEADS * SEQ * HEAD_DIM; // 4096*2048

    {
        int n;
        n = SEQ * HID;
        cvt_bf16_kernel<<<(n + 255) / 256, 256, 0, stream>>>(hidden, Abf, n);
        n = QKV_OUT * HID;
        cvt_bf16_kernel<<<(n + 255) / 256, 256, 0, stream>>>(w_qkv, Wq, n);
        n = HID * HID;
        cvt_bf16_kernel<<<(n + 255) / 256, 256, 0, stream>>>(w_o, Wo, n);
    }

    qkv_gemm_kernel<<<dim3(SEQ / 64, QKV_OUT / 128), 256, 0, stream>>>(
        Abf, Wq, qb, kb, vT);

    {
        int n = SEQ * NUM_HEADS * (HEAD_DIM / 2);
        rope_kernel<<<(n + 255) / 256, 256, 0, stream>>>(qb, kb);
    }

    // 16 heads x 32 q-blocks of 128 rows
    attn_kernel<<<dim3(NUM_HEADS * (SEQ / 128)), 256, 0, stream>>>(
        qb, kb, vT, atn);

    out_gemm_kernel<<<dim3(SEQ / 64, HID / 128), 256, 0, stream>>>(atn, Wo, out);
}
